// TestDecoderLayer_7645041787400
// MI455X (gfx1250) — compile-verified, matrix operand order checked by reading, not run on hardware
//
#include <hip/hip_runtime.h>
#include <hip/hip_bf16.h>
#include <cstdint>

// ---- problem dims (B=1) ----
#define T_TOK 1024
#define HDIM  2048
#define NHEADS 32
#define NKVH   8
#define DHEAD  64
#define NEXP   8
#define FDIM   8192

typedef __attribute__((ext_vector_type(16))) __bf16 v16bf;
typedef __attribute__((ext_vector_type(8)))  __bf16 v8bf;
typedef __attribute__((ext_vector_type(8)))  float  v8f;
typedef __bf16 bf16_t;

// native converts: v_cvt_*bf16_f32 on gfx1250
__device__ inline bf16_t f2bf(float x) { return (bf16_t)x; }

__device__ inline v8f zero8() {
  v8f z;
#pragma unroll
  for (int i = 0; i < 8; ++i) z[i] = 0.f;
  return z;
}

__device__ inline v16bf concat8(v8bf lo, v8bf hi) {
  return __builtin_shufflevector(lo, hi, 0, 1, 2, 3, 4, 5, 6, 7,
                                 8, 9, 10, 11, 12, 13, 14, 15);
}

__device__ inline v8f wmma_bf16(v16bf a, v16bf b, v8f c) {
  // D = A(16x32 bf16) * B(32x16 bf16) + C(16x16 f32)
  return __builtin_amdgcn_wmma_f32_16x16x32_bf16(false, a, false, b, (short)0, c,
                                                 false, false);
}

// A-fragment (16-bit A 16x32 layout): lane holds row (lane&15); half-select
// (lane>>4) picks K-octets {hs*8..+7} and {16+hs*8..+7} -> two b128 loads.
__device__ inline v16bf load_a_bf(const bf16_t* A, int lda, int row0, int k0) {
  int lane = threadIdx.x & 31;
  int r  = row0 + (lane & 15);
  int hs = lane >> 4;
  const bf16_t* p = A + (size_t)r * lda + k0 + hs * 8;
  v8bf lo = *(const v8bf*)p;
  v8bf hi = *(const v8bf*)(p + 16);
  return concat8(lo, hi);
}

// Same, but rows gathered through an index list (MoE token gather).
__device__ inline v16bf load_a_gather(const bf16_t* A, int lda, const int* rows,
                                      int row0, int k0) {
  int lane = threadIdx.x & 31;
  int r  = rows[row0 + (lane & 15)];
  int hs = lane >> 4;
  const bf16_t* p = A + (size_t)r * lda + k0 + hs * 8;
  v8bf lo = *(const v8bf*)p;
  v8bf hi = *(const v8bf*)(p + 16);
  return concat8(lo, hi);
}

// B-fragment from row-major f32 weights (KxN), converted bf16 in-register.
// lane holds column (lane&15); K pattern mirrors the A layout.
__device__ inline v16bf load_b_f32(const float* B, int ldb, int k0, int col0) {
  int lane = threadIdx.x & 31;
  int c  = col0 + (lane & 15);
  int hs = lane >> 4;
  const float* p = B + (size_t)(k0 + hs * 8) * ldb + c;
  v16bf b;
#pragma unroll
  for (int j = 0; j < 8; ++j) b[j] = f2bf(p[(size_t)j * ldb]);
#pragma unroll
  for (int j = 0; j < 8; ++j) b[8 + j] = f2bf(p[(size_t)(16 + j) * ldb]);
  return b;
}

// ---------------- rmsnorm: y = x * rsqrt(mean(x^2)+eps) * w ----------------
__global__ __launch_bounds__(256) void rmsnorm_k(const float* __restrict__ x,
                                                 const float* __restrict__ w,
                                                 float* yf, bf16_t* yb) {
  int row = blockIdx.x;
  const float* xr = x + (size_t)row * HDIM;
  float s = 0.f;
  for (int i = threadIdx.x; i < HDIM; i += 256) { float v = xr[i]; s += v * v; }
  __shared__ float red[256];
  red[threadIdx.x] = s;
  __syncthreads();
  for (int st = 128; st > 0; st >>= 1) {
    if (threadIdx.x < st) red[threadIdx.x] += red[threadIdx.x + st];
    __syncthreads();
  }
  float inv = rsqrtf(red[0] / (float)HDIM + 1e-5f);
  for (int i = threadIdx.x; i < HDIM; i += 256) {
    float v = xr[i] * inv * w[i];
    if (yf) yf[(size_t)row * HDIM + i] = v;
    yb[(size_t)row * HDIM + i] = f2bf(v);
  }
}

// ---- generic GEMM: C[M,N] = A_bf16[M,K] @ Bf32[K,N] (+resid), 128x16/wave --
// one wave computes a 128x16 macro-tile (8 m-subtiles reuse one B-fragment:
// 2 B-dwords per WMMA); m-fastest tile order keeps B resident in L2.
template <bool RES, bool OUTF, bool OUTB>
__global__ __launch_bounds__(256) void gemm_bf16w(const bf16_t* __restrict__ A,
                                                  const float* __restrict__ Bw,
                                                  float* __restrict__ Cf,
                                                  bf16_t* __restrict__ Cb,
                                                  const float* __restrict__ resid,
                                                  int M, int N, int K) {
  int wave = blockIdx.x * 8 + (threadIdx.x >> 5);
  int m128 = M >> 7;
  int tm = wave % m128;
  int tn = wave / m128;
  if (tn >= (N >> 4)) return;
  int row0 = tm << 7;
  v8f acc[8];
#pragma unroll
  for (int s = 0; s < 8; ++s) acc[s] = zero8();
  for (int k0 = 0; k0 < K; k0 += 32) {
    v16bf b = load_b_f32(Bw, N, k0, tn * 16);
#pragma unroll
    for (int s = 0; s < 8; ++s) {
      v16bf a = load_a_bf(A, K, row0 + s * 16, k0);
      acc[s] = wmma_bf16(a, b, acc[s]);
    }
  }
  int lane = threadIdx.x & 31;
  int n = tn * 16 + (lane & 15);
#pragma unroll
  for (int s = 0; s < 8; ++s) {
    int mbase = row0 + s * 16 + ((lane >> 4) << 3);
#pragma unroll
    for (int i = 0; i < 8; ++i) {
      size_t idx = (size_t)(mbase + i) * N + n;
      float v = acc[s][i];
      if constexpr (RES) v += resid[idx];
      if constexpr (OUTF) Cf[idx] = v;
      if constexpr (OUTB) Cb[idx] = f2bf(v);
    }
  }
}

// ---------------- flash attention, one wave per (head, 16-query block) -----
__global__ __launch_bounds__(32) void attn_k(const bf16_t* __restrict__ Qb,
                                             const bf16_t* __restrict__ Kb,
                                             const bf16_t* __restrict__ Vb,
                                             bf16_t* __restrict__ Ob) {
  const int qb   = blockIdx.x & 63;   // S/16 = 64 query blocks
  const int head = blockIdx.x >> 6;   // 32 heads
  const int kvh  = head >> 2;         // GROUPS = 4
  const int lane = threadIdx.x;
  const int ln = lane & 15, hs = lane >> 4;
  const int ldkv = NKVH * DHEAD;      // 512

  __shared__ float ptile[16][17];

  v16bf aQ0 = load_a_bf(Qb, NHEADS * DHEAD, qb * 16, head * DHEAD);
  v16bf aQ1 = load_a_bf(Qb, NHEADS * DHEAD, qb * 16, head * DHEAD + 32);

  v8f o[4] = {zero8(), zero8(), zero8(), zero8()};
  float rm[8], rl[8];
#pragma unroll
  for (int i = 0; i < 8; ++i) { rm[i] = -3.0e38f; rl[i] = 0.f; }

  for (int kb = 0; kb <= qb; ++kb) {
    // K^T B-fragments: col = key, contiguous along d within a row of K
    const bf16_t* kp = Kb + (size_t)(kb * 16 + ln) * ldkv + kvh * DHEAD + hs * 8;
    v16bf bK0 = concat8(*(const v8bf*)kp,        *(const v8bf*)(kp + 16));
    v16bf bK1 = concat8(*(const v8bf*)(kp + 32), *(const v8bf*)(kp + 48));

    v8f s = zero8();
    s = wmma_bf16(aQ0, bK0, s);
    s = wmma_bf16(aQ1, bK1, s);

    float p[8];
#pragma unroll
    for (int i = 0; i < 8; ++i) {
      float v = s[i] * 0.125f;                 // 1/sqrt(64)
      int m = i + (hs << 3);
      if (kb == qb && ln > m) v = -3.0e38f;    // causal
      p[i] = v;
    }
    // online softmax: row stats across the 16 lanes of each half
    float scale[8];
#pragma unroll
    for (int i = 0; i < 8; ++i) {
      float mx = p[i];
      for (int d = 1; d < 16; d <<= 1) mx = fmaxf(mx, __shfl_xor(mx, d, 32));
      float mnew = fmaxf(rm[i], mx);
      float sc = __expf(rm[i] - mnew);
      float pe = __expf(p[i] - mnew);
      float ps = pe;
      for (int d = 1; d < 16; d <<= 1) ps += __shfl_xor(ps, d, 32);
      rl[i] = rl[i] * sc + ps;
      rm[i] = mnew;
      scale[i] = sc;
      p[i] = pe;
    }
#pragma unroll
    for (int t = 0; t < 4; ++t)
#pragma unroll
      for (int i = 0; i < 8; ++i) o[t][i] *= scale[i];

    // transpose P (C-layout -> A-layout) through LDS
#pragma unroll
    for (int i = 0; i < 8; ++i) ptile[i + (hs << 3)][ln] = p[i];
    asm volatile("s_wait_dscnt 0x0" ::: "memory");
    v16bf aP;
#pragma unroll
    for (int j = 0; j < 8; ++j) aP[j] = f2bf(ptile[ln][(hs << 3) + j]);
#pragma unroll
    for (int j = 0; j < 8; ++j) aP[8 + j] = f2bf(0.f);  // keys 16..31 unused
    asm volatile("s_wait_dscnt 0x0" ::: "memory");

    // PV: 4 d-tiles of 16 columns; K rows 16..31 zero-padded
#pragma unroll
    for (int t = 0; t < 4; ++t) {
      v16bf bV;
#pragma unroll
      for (int j = 0; j < 8; ++j)
        bV[j] = Vb[(size_t)(kb * 16 + (hs << 3) + j) * ldkv + kvh * DHEAD + t * 16 + ln];
#pragma unroll
      for (int j = 0; j < 8; ++j) bV[8 + j] = f2bf(0.f);
      o[t] = wmma_bf16(aP, bV, o[t]);
    }
  }

#pragma unroll
  for (int i = 0; i < 8; ++i) {
    float inv = 1.0f / rl[i];
    int row = qb * 16 + i + (hs << 3);
    bf16_t* orow = Ob + (size_t)row * (NHEADS * DHEAD) + head * DHEAD + ln;
    orow[0]  = f2bf(o[0][i] * inv);
    orow[16] = f2bf(o[1][i] * inv);
    orow[32] = f2bf(o[2][i] * inv);
    orow[48] = f2bf(o[3][i] * inv);
  }
}

// ---------------- gating + top-2 routing ----------------
__global__ void route_init_k(int* cnt, int* tlist, float* twgt) {
  int i = blockIdx.x * 256 + threadIdx.x;
  if (i < NEXP) cnt[i] = 0;
  if (i < NEXP * T_TOK) { tlist[i] = 0; twgt[i] = 0.f; }
}

__global__ __launch_bounds__(256) void gate_topk_k(const float* __restrict__ h2n,
                                                   const float* __restrict__ gw,
                                                   int* cnt, int* tlist, float* twgt) {
  int t = blockIdx.x;
  int e = threadIdx.x >> 5;        // 8 waves, one expert each
  int lane = threadIdx.x & 31;
  float acc = 0.f;
  for (int i = lane; i < HDIM; i += 32)
    acc += h2n[(size_t)t * HDIM + i] * gw[(size_t)i * NEXP + e];
  for (int d = 1; d < 32; d <<= 1) acc += __shfl_xor(acc, d, 32);
  __shared__ float logits[NEXP];
  if (lane == 0) logits[e] = acc;
  __syncthreads();
  if (threadIdx.x == 0) {
    float p[NEXP];
    float mx = logits[0];
    for (int i = 1; i < NEXP; ++i) mx = fmaxf(mx, logits[i]);
    float se = 0.f;
    for (int i = 0; i < NEXP; ++i) { p[i] = __expf(logits[i] - mx); se += p[i]; }
    for (int i = 0; i < NEXP; ++i) p[i] /= se;
    int i0 = 0;
    for (int i = 1; i < NEXP; ++i) if (p[i] > p[i0]) i0 = i;
    int i1 = (i0 == 0) ? 1 : 0;
    for (int i = 0; i < NEXP; ++i) if (i != i0 && p[i] > p[i1]) i1 = i;
    float s2 = p[i0] + p[i1];
    int pos = atomicAdd(&cnt[i0], 1);
    tlist[i0 * T_TOK + pos] = t; twgt[i0 * T_TOK + pos] = p[i0] / s2;
    pos = atomicAdd(&cnt[i1], 1);
    tlist[i1 * T_TOK + pos] = t; twgt[i1 * T_TOK + pos] = p[i1] / s2;
  }
}

// ---- MoE up: gated = relu(gather(t)@w1[e]) * (gather(t)@w3[e]) -> bf16 ----
// 128x16 macro-tile per wave; one A-octet feeds both w1 and w3 fragments
// (2 B-dwords per WMMA)
__global__ __launch_bounds__(256) void moe_up_k(int e, const bf16_t* __restrict__ A,
                                                const float* __restrict__ w1,
                                                const float* __restrict__ w3,
                                                const int* __restrict__ tlist,
                                                const int* __restrict__ cnt,
                                                bf16_t* __restrict__ gbuf) {
  int wave = blockIdx.x * 8 + (threadIdx.x >> 5);
  const int m128 = T_TOK >> 7;
  int tm = wave % m128;
  int tn = wave / m128;
  if (tn >= (FDIM >> 4)) return;
  int ce = cnt[e];
  int row0 = tm << 7;
  if (row0 >= ce) return;
  const int* rows = tlist + e * T_TOK;
  const float* B1 = w1 + (size_t)e * HDIM * FDIM;
  const float* B3 = w3 + (size_t)e * HDIM * FDIM;
  v8f a1[8], a3[8];
#pragma unroll
  for (int s = 0; s < 8; ++s) { a1[s] = zero8(); a3[s] = zero8(); }
  for (int k0 = 0; k0 < HDIM; k0 += 32) {
    v16bf b1 = load_b_f32(B1, FDIM, k0, tn * 16);
    v16bf b3 = load_b_f32(B3, FDIM, k0, tn * 16);
#pragma unroll
    for (int s = 0; s < 8; ++s) {
      v16bf a = load_a_gather(A, HDIM, rows, row0 + s * 16, k0);
      a1[s] = wmma_bf16(a, b1, a1[s]);
      a3[s] = wmma_bf16(a, b3, a3[s]);
    }
  }
  int lane = threadIdx.x & 31;
  int n = tn * 16 + (lane & 15);
#pragma unroll
  for (int s = 0; s < 8; ++s) {
    int mbase = row0 + s * 16 + ((lane >> 4) << 3);
#pragma unroll
    for (int i = 0; i < 8; ++i) {
      float u = a1[s][i];
      u = (u > 0.f) ? u : 0.f;
      gbuf[(size_t)(mbase + i) * FDIM + n] = f2bf(u * a3[s][i]);
    }
  }
}

// ---- MoE down: out[tok] += wgt * (gated @ w2[e]); race-free non-atomic ----
__global__ __launch_bounds__(256) void moe_down_k(int e, const bf16_t* __restrict__ gbuf,
                                                  const float* __restrict__ w2,
                                                  const int* __restrict__ tlist,
                                                  const float* __restrict__ twgt,
                                                  const int* __restrict__ cnt,
                                                  float* __restrict__ out) {
  int wave = blockIdx.x * 8 + (threadIdx.x >> 5);
  const int m128 = T_TOK >> 7;
  int tm = wave % m128;
  int tn = wave / m128;
  if (tn >= (HDIM >> 4)) return;
  int ce = cnt[e];
  int row0 = tm << 7;
  if (row0 >= ce) return;
  const float* B = w2 + (size_t)e * FDIM * HDIM;
  v8f acc[8];
#pragma unroll
  for (int s = 0; s < 8; ++s) acc[s] = zero8();
  for (int k0 = 0; k0 < FDIM; k0 += 32) {
    v16bf b = load_b_f32(B, HDIM, k0, tn * 16);
#pragma unroll
    for (int s = 0; s < 8; ++s) {
      v16bf a = load_a_bf(gbuf, FDIM, row0 + s * 16, k0);
      acc[s] = wmma_bf16(a, b, acc[s]);
    }
  }
  int lane = threadIdx.x & 31;
  int n = tn * 16 + (lane & 15);
#pragma unroll
  for (int s = 0; s < 8; ++s) {
    int mbase = row0 + s * 16 + ((lane >> 4) << 3);
#pragma unroll
    for (int i = 0; i < 8; ++i) {
      int m = mbase + i;
      if (m < ce) {
        int tok = tlist[e * T_TOK + m];
        float w = twgt[e * T_TOK + m];
        out[(size_t)tok * HDIM + n] += w * acc[s][i];  // unique (tok,e); serialized
      }
    }
  }
}

extern "C" void kernel_launch(void* const* d_in, const int* in_sizes, int n_in,
                              void* d_out, int out_size, void* d_ws, size_t ws_size,
                              hipStream_t stream) {
  (void)in_sizes; (void)n_in; (void)out_size; (void)ws_size;
  const float* x      = (const float*)d_in[0];
  const float* ln1_w  = (const float*)d_in[1];
  const float* wq     = (const float*)d_in[2];
  const float* wk     = (const float*)d_in[3];
  const float* wv     = (const float*)d_in[4];
  const float* wo     = (const float*)d_in[5];
  const float* ln2_w  = (const float*)d_in[6];
  const float* gate_w = (const float*)d_in[7];
  const float* w1     = (const float*)d_in[8];
  const float* w2     = (const float*)d_in[9];
  const float* w3     = (const float*)d_in[10];
  float* out = (float*)d_out;

  char* ws = (char*)d_ws;
  size_t off = 0;
  auto alloc = [&](size_t bytes) -> void* {
    void* p = ws + off;
    off = (off + bytes + 255) & ~(size_t)255;
    return p;
  };
  bf16_t* h1n_b = (bf16_t*)alloc((size_t)T_TOK * HDIM * 2);
  bf16_t* qbuf  = (bf16_t*)alloc((size_t)T_TOK * NHEADS * DHEAD * 2);
  bf16_t* kbuf  = (bf16_t*)alloc((size_t)T_TOK * NKVH * DHEAD * 2);
  bf16_t* vbuf  = (bf16_t*)alloc((size_t)T_TOK * NKVH * DHEAD * 2);
  bf16_t* attnb = (bf16_t*)alloc((size_t)T_TOK * NHEADS * DHEAD * 2);
  float*  hbuf  = (float*) alloc((size_t)T_TOK * HDIM * 4);
  float*  h2n_f = (float*) alloc((size_t)T_TOK * HDIM * 4);
  bf16_t* h2n_b = (bf16_t*)alloc((size_t)T_TOK * HDIM * 2);
  int*    cnt   = (int*)   alloc(NEXP * 4);
  int*    tlist = (int*)   alloc((size_t)NEXP * T_TOK * 4);
  float*  twgt  = (float*) alloc((size_t)NEXP * T_TOK * 4);
  bf16_t* gbuf  = (bf16_t*)alloc((size_t)T_TOK * FDIM * 2);  // reused per expert

  const dim3 b256(256), b32(32);
  auto gemm_blocks = [](int M, int N) { return ((M >> 7) * (N >> 4) + 7) / 8; };

  // h1n = rmsnorm(x, ln1_w) (bf16)
  rmsnorm_k<<<T_TOK, b256, 0, stream>>>(x, ln1_w, nullptr, h1n_b);
  // q/k/v projections
  gemm_bf16w<false, false, true><<<gemm_blocks(T_TOK, NHEADS * DHEAD), b256, 0, stream>>>(
      h1n_b, wq, nullptr, qbuf, nullptr, T_TOK, NHEADS * DHEAD, HDIM);
  gemm_bf16w<false, false, true><<<gemm_blocks(T_TOK, NKVH * DHEAD), b256, 0, stream>>>(
      h1n_b, wk, nullptr, kbuf, nullptr, T_TOK, NKVH * DHEAD, HDIM);
  gemm_bf16w<false, false, true><<<gemm_blocks(T_TOK, NKVH * DHEAD), b256, 0, stream>>>(
      h1n_b, wv, nullptr, vbuf, nullptr, T_TOK, NKVH * DHEAD, HDIM);
  // flash attention (GQA 4:1, causal)
  attn_k<<<(T_TOK / 16) * NHEADS, b32, 0, stream>>>(qbuf, kbuf, vbuf, attnb);
  // h = x + attn @ wo
  gemm_bf16w<true, true, false><<<gemm_blocks(T_TOK, HDIM), b256, 0, stream>>>(
      attnb, wo, hbuf, nullptr, x, T_TOK, HDIM, HDIM);
  // h2n = rmsnorm(h, ln2_w) (f32 for gate, bf16 for GEMMs)
  rmsnorm_k<<<T_TOK, b256, 0, stream>>>(hbuf, ln2_w, h2n_f, h2n_b);
  // routing
  route_init_k<<<(NEXP * T_TOK + 255) / 256, b256, 0, stream>>>(cnt, tlist, twgt);
  gate_topk_k<<<T_TOK, b256, 0, stream>>>(h2n_f, gate_w, cnt, tlist, twgt);
  // out = h, then MoE scatters on top
  hipMemcpyAsync(out, hbuf, (size_t)T_TOK * HDIM * 4, hipMemcpyDeviceToDevice, stream);
  // per-expert grouped GEMMs (sequential launches reuse gbuf, serialize scatter)
  const int up_blocks   = ((T_TOK >> 7) * (FDIM >> 4) + 7) / 8;
  const int down_blocks = ((T_TOK >> 7) * (HDIM >> 4) + 7) / 8;
  for (int e = 0; e < NEXP; ++e) {
    moe_up_k<<<up_blocks, b256, 0, stream>>>(e, h2n_b, w1, w3, tlist, cnt, gbuf);
    moe_down_k<<<down_blocks, b256, 0, stream>>>(e, gbuf, w2, tlist, twgt, cnt, out);
  }
}
